// RNNLayer_85194971283753
// MI455X (gfx1250) — matrix operand order repaired and checked by reading, not compiled, not run on hardware
//
#include <hip/hip_runtime.h>

// Fused LSTM: B=256, T=512, I=64, H=128, 4H=512.
// 16 workgroups x 8 wave32s. Each wave owns 16 h-columns across ALL FOUR
// gates (n-tiles w, w+8, w+16, w+24): i/f/g/o for an (m,c) land in the same
// lane in 4 accumulators -> register-only cell update (no gates LDS).
// FULLY weights-stationary: both W_ih and W_hh B-fragments live in VGPRs for
// all 512 steps (~300 VGPRs, using gfx1250's >256 VGPR-MSB addressing).
// h state double-buffered bf16 in LDS (8 KB total) -> ONE barrier per step.
// c state in VGPRs. x pre-converted/pre-swizzled to bf16 A-fragment order.

#define B_  256
#define T_  512
#define I_  64
#define H_  128
#define G_  512
#define ROWS 16            // batch rows per workgroup (WMMA M)
#define NWAVES 8           // 256 threads

typedef __attribute__((ext_vector_type(16))) __bf16 v16bf;
typedef __attribute__((ext_vector_type(8)))  __bf16 v8bf;
typedef __attribute__((ext_vector_type(8)))  float  v8f;

union AFrag { v16bf v; v8bf h[2]; unsigned short u[16]; };

__device__ __forceinline__ unsigned short f2bf(float f) {
    return (unsigned short)(__float_as_uint(f) >> 16);
}
__device__ __forceinline__ float sigm_f(float x) { return 1.0f / (1.0f + __expf(-x)); }
__device__ __forceinline__ float tanh_f(float x) { return 2.0f / (1.0f + __expf(-2.0f * x)) - 1.0f; }

// ---- prepass: convert x to bf16, pre-swizzled into A-fragment element order.
// Per (b,t) row of 64: pos p -> half=p>>5, kt=(p>>4)&1, j=p&15,
// k = kt*32 + (j<8 ? half*8+j : 16+half*8+(j-8)).  Lane then loads 32 B flat.
__global__ void __launch_bounds__(256)
x_to_bf16_kernel(const float* __restrict__ x, unsigned short* __restrict__ xbf) {
    const size_t i = (size_t)blockIdx.x * blockDim.x + threadIdx.x;
    const int p = (int)(i & 63);
    const size_t bt = i >> 6;
    const int hf = p >> 5, kt = (p >> 4) & 1, j = p & 15;
    const int k = kt * 32 + (j < 8 ? hf * 8 + j : 16 + hf * 8 + (j - 8));
    xbf[bt * 64 + p] = f2bf(x[bt * 64 + k]);
}

template <bool PRE>
__global__ void __launch_bounds__(NWAVES * 32)
lstm_fused_kernel(const float* __restrict__ inputs,          // [B,T,I] f32
                  const unsigned short* __restrict__ xbf,    // [B,T,64] bf16 frag-order (PRE)
                  const float* __restrict__ h_prev,
                  const float* __restrict__ c_prev,
                  const float* __restrict__ W_ih,            // [4H,I]
                  const float* __restrict__ W_hh,            // [4H,H]
                  const float* __restrict__ b_ih,
                  const float* __restrict__ b_hh,
                  float* __restrict__ out)                   // [B*T*H | B*H | B*H]
{
    __shared__ unsigned short hb[2][ROWS * H_];   // double-buffered bf16 h: 8 KB

    const int tid  = threadIdx.x;
    const int wave = tid >> 5;
    const int lane = tid & 31;
    const int b0   = blockIdx.x * ROWS;
    const int am   = lane & 15;          // A row (M) / column-in-tile
    const int half = (lane >> 4) & 1;    // K-half selector
    const int mb   = half * 8;           // C-fragment row base for this lane

    // ---- ALL weight B-fragments VGPR-resident: gate g -> n-tile (g*8 + wave)
    // B 32x16 layout: lane half h holds K=8h..8h+7 then K=16+8h..16+8h+7 of col N=lane%16.
    v16bf wih[4][2];                     // x-projection, K=64
    v16bf whh[4][4];                     // h-projection, K=128
    #pragma unroll
    for (int g = 0; g < 4; ++g) {
        const int n = g * 128 + wave * 16 + am;   // global gate column
        #pragma unroll
        for (int kt = 0; kt < 2; ++kt) {
            const float* src = W_ih + (size_t)n * I_ + kt * 32 + half * 8;
            AFrag tmp;
            #pragma unroll
            for (int j = 0; j < 16; ++j) tmp.u[j] = f2bf(src[j < 8 ? j : j + 8]);
            wih[g][kt] = tmp.v;
        }
        #pragma unroll
        for (int kt = 0; kt < 4; ++kt) {
            const float* src = W_hh + (size_t)n * H_ + kt * 32 + half * 8;
            AFrag tmp;
            #pragma unroll
            for (int j = 0; j < 16; ++j) tmp.u[j] = f2bf(src[j < 8 ? j : j + 8]);
            whh[g][kt] = tmp.v;
        }
    }
    // ---- h state (bf16, buffer 0) ----
    for (int i = tid; i < ROWS * H_; i += blockDim.x) {
        int m = i >> 7, c = i & (H_ - 1);
        hb[0][i] = f2bf(h_prev[(size_t)(b0 + m) * H_ + c]);
    }
    // ---- c state + bias in registers (same distribution as accumulators) ----
    const int ccol = wave * 16 + am;     // this lane's h column
    float creg[8];
    #pragma unroll
    for (int r = 0; r < 8; ++r)
        creg[r] = c_prev[(size_t)(b0 + mb + r) * H_ + ccol];
    float biasv[4];
    #pragma unroll
    for (int g = 0; g < 4; ++g) {
        const int n = g * 128 + ccol;
        biasv[g] = b_ih[n] + b_hh[n];
    }
    __syncthreads();

    const unsigned short* xrow_base = PRE ? (xbf + (size_t)(b0 + am) * (T_ * I_)) : nullptr;
    const float*          frow_base = inputs + (size_t)(b0 + am) * (T_ * I_);
    float* hout = out + (size_t)B_ * T_ * H_;
    float* cout = hout + (size_t)B_ * H_;

    int p = 0;
    for (int t = 0; t < T_; ++t) {
        // ---- h A-fragments from LDS (8x ds_load_b128) ----
        AFrag ha[4];
        const unsigned short* hp = hb[p] + am * H_;
        #pragma unroll
        for (int kt = 0; kt < 4; ++kt) {
            ha[kt].h[0] = *(const v8bf*)(hp + kt * 32 + half * 8);
            ha[kt].h[1] = *(const v8bf*)(hp + kt * 32 + 16 + half * 8);
        }

        // ---- x A-fragments ----
        AFrag xa[2];
        if (PRE) {
            const unsigned short* xr = xrow_base + (size_t)t * I_;
            if (t + 1 < T_) __builtin_prefetch(xr + I_, 0, 0);
            xa[0].v = *(const v16bf*)(xr + half * 32);
            xa[1].v = *(const v16bf*)(xr + half * 32 + 16);
        } else {
            const float* xr = frow_base + (size_t)t * I_;
            if (t + 1 < T_) __builtin_prefetch(xr + I_, 0, 0);
            #pragma unroll
            for (int kt = 0; kt < 2; ++kt) {
                const float4 a0 = *(const float4*)(xr + kt * 32 + half * 8);
                const float4 a1 = *(const float4*)(xr + kt * 32 + half * 8 + 4);
                const float4 c0 = *(const float4*)(xr + kt * 32 + 16 + half * 8);
                const float4 c1 = *(const float4*)(xr + kt * 32 + 16 + half * 8 + 4);
                xa[kt].u[0]=f2bf(a0.x); xa[kt].u[1]=f2bf(a0.y); xa[kt].u[2]=f2bf(a0.z); xa[kt].u[3]=f2bf(a0.w);
                xa[kt].u[4]=f2bf(a1.x); xa[kt].u[5]=f2bf(a1.y); xa[kt].u[6]=f2bf(a1.z); xa[kt].u[7]=f2bf(a1.w);
                xa[kt].u[8]=f2bf(c0.x); xa[kt].u[9]=f2bf(c0.y); xa[kt].u[10]=f2bf(c0.z); xa[kt].u[11]=f2bf(c0.w);
                xa[kt].u[12]=f2bf(c1.x); xa[kt].u[13]=f2bf(c1.y); xa[kt].u[14]=f2bf(c1.z); xa[kt].u[15]=f2bf(c1.w);
            }
        }

        // ---- 4 gate chains (i,f,g,o), same 16 cols; all B operands in VGPRs ----
        v8f acc[4];
        #pragma unroll
        for (int g = 0; g < 4; ++g) {
            v8f a;
            #pragma unroll
            for (int r = 0; r < 8; ++r) a[r] = biasv[g];
            a = __builtin_amdgcn_wmma_f32_16x16x32_bf16(false, xa[0].v, false, wih[g][0], (short)0, a, false, false);
            a = __builtin_amdgcn_wmma_f32_16x16x32_bf16(false, xa[1].v, false, wih[g][1], (short)0, a, false, false);
            a = __builtin_amdgcn_wmma_f32_16x16x32_bf16(false, ha[0].v, false, whh[g][0], (short)0, a, false, false);
            a = __builtin_amdgcn_wmma_f32_16x16x32_bf16(false, ha[1].v, false, whh[g][1], (short)0, a, false, false);
            a = __builtin_amdgcn_wmma_f32_16x16x32_bf16(false, ha[2].v, false, whh[g][2], (short)0, a, false, false);
            a = __builtin_amdgcn_wmma_f32_16x16x32_bf16(false, ha[3].v, false, whh[g][3], (short)0, a, false, false);
            acc[g] = a;
        }

        // ---- register-only cell update: lane holds i/f/g/o for (mb+r, ccol) ----
        unsigned short* hdst = hb[p ^ 1];
        #pragma unroll
        for (int r = 0; r < 8; ++r) {
            const int m = mb + r;
            const float ig = sigm_f(acc[0][r]);
            const float fg = sigm_f(acc[1][r]);
            const float gg = tanh_f(acc[2][r]);
            const float og = sigm_f(acc[3][r]);
            const float cn = fg * creg[r] + ig * gg;
            creg[r] = cn;
            const float hn = og * tanh_f(cn);
            out[(size_t)(b0 + m) * (T_ * H_) + (size_t)t * H_ + ccol] = hn;
            hdst[m * H_ + ccol] = f2bf(hn);
            if (t == T_ - 1) {
                hout[(size_t)(b0 + m) * H_ + ccol] = hn;   // h_last
                cout[(size_t)(b0 + m) * H_ + ccol] = cn;   // c_last
            }
        }
        __syncthreads();                 // single barrier: hdst complete
        p ^= 1;
    }
}

extern "C" void kernel_launch(void* const* d_in, const int* in_sizes, int n_in,
                              void* d_out, int out_size, void* d_ws, size_t ws_size,
                              hipStream_t stream) {
    (void)in_sizes; (void)n_in; (void)out_size;
    const float* inputs = (const float*)d_in[0];
    const float* h_prev = (const float*)d_in[1];
    const float* c_prev = (const float*)d_in[2];
    const float* W_ih   = (const float*)d_in[3];
    const float* W_hh   = (const float*)d_in[4];
    const float* b_ih   = (const float*)d_in[5];
    const float* b_hh   = (const float*)d_in[6];
    float* out = (float*)d_out;

    const size_t xbf_bytes = (size_t)B_ * T_ * I_ * sizeof(unsigned short);
    const bool pre = (d_ws != nullptr) && (ws_size >= xbf_bytes);

    dim3 grid(B_ / ROWS), block(NWAVES * 32);
    if (pre) {
        unsigned short* xbf = (unsigned short*)d_ws;
        const size_t nel = (size_t)B_ * T_ * I_;
        x_to_bf16_kernel<<<(unsigned)(nel / 256), 256, 0, stream>>>(inputs, xbf);
        lstm_fused_kernel<true><<<grid, block, 0, stream>>>(
            inputs, xbf, h_prev, c_prev, W_ih, W_hh, b_ih, b_hh, out);
    } else {
        lstm_fused_kernel<false><<<grid, block, 0, stream>>>(
            inputs, nullptr, h_prev, c_prev, W_ih, W_hh, b_ih, b_hh, out);
    }
}
